// RetinaNetDetectorModel_23441931502258
// MI455X (gfx1250) — compile-verified, matrix operand order checked by reading, not run on hardware
//
#include <hip/hip_runtime.h>
#include <stdint.h>

// ---------------- problem constants (match reference) ----------------
#define N_CLASSES   8
#define TOPK        1000
#define KPAD        1024          // padded per-class candidate count (power of 2)
#define DET         300
#define CAND_CAP    2048
#define NMS_TH      0.5f
#define SCORE_TH    0.05f
#define BBOX_CLIP   4.135166556742356f   // log(1000/16)

// ---------------- CDNA5 WMMA types (probe-verified signatures) -------
typedef __bf16 bf16_t;
typedef bf16_t v16bf __attribute__((ext_vector_type(16)));
typedef float  v8f   __attribute__((ext_vector_type(8)));

// ---------------- workspace layout (bytes) ---------------------------
#define WS_HIST1    0              // 8*256 u32
#define WS_HIST2    8192           // 8*256 u32
#define WS_B1       16384          // 8 u32
#define WS_ABOVE1   16416          // 8 u32
#define WS_THRESH   16448          // 8 u32
#define WS_CNT      16480          // 8 u32
#define WS_CAND     32768          // 8*2048 u64  (131072 B)
#define WS_SELIDX   163840         // 8*1024 u32
#define WS_SELSC    196608         // 8*1024 f32
#define WS_BOXES    229376         // 8*1024*6 f32
#define WS_VOL      425984         // 8*1024 f32
#define WS_FSC      458752         // 8*1024 f32
#define WS_MASK     491520         // 8*1024*32 u32 (1 MB)
#define WS_ZERO1_BYTES 32768
#define WS_MASK_BYTES  (N_CLASSES * KPAD * 32 * 4)

// order-preserving float->u32 key (unsigned compare == float compare)
__device__ __forceinline__ uint32_t fkey(float f) {
  uint32_t u = __float_as_uint(f);
  return (u & 0x80000000u) ? ~u : (u | 0x80000000u);
}
__device__ __forceinline__ float unkey(uint32_t k) {
  uint32_t u = (k & 0x80000000u) ? (k & 0x7fffffffu) : ~k;
  return __uint_as_float(u);
}

// ============= pass 1/2: two-level 256-bin histograms on logit keys ===
__global__ void k_hist(const float* __restrict__ logits, uint32_t* __restrict__ ghist,
                       const uint32_t* __restrict__ b1, int pass, long long total) {
  __shared__ uint32_t lh[N_CLASSES * 256];   // 8 KB, LDS-privatized
  for (int t = threadIdx.x; t < N_CLASSES * 256; t += blockDim.x) lh[t] = 0u;
  __syncthreads();
  long long stride = (long long)gridDim.x * blockDim.x;
  for (long long e = (long long)blockIdx.x * blockDim.x + threadIdx.x; e < total; e += stride) {
    __builtin_prefetch(logits + e + stride, 0, 1);   // -> global_prefetch_b8
    float f = logits[e];
    int c = (int)(e % N_CLASSES);      // layout [N, C], C contiguous
    uint32_t k = fkey(f);
    if (pass == 0) {
      atomicAdd(&lh[c * 256 + (k >> 24)], 1u);
    } else {
      if ((k >> 24) == b1[c]) atomicAdd(&lh[c * 256 + ((k >> 16) & 255u)], 1u);
    }
  }
  __syncthreads();
  for (int t = threadIdx.x; t < N_CLASSES * 256; t += blockDim.x)
    if (lh[t]) atomicAdd(&ghist[t], lh[t]);
}

__global__ void k_scan1(const uint32_t* __restrict__ gh, uint32_t* __restrict__ b1,
                        uint32_t* __restrict__ above1) {
  int c = blockIdx.x;
  if (threadIdx.x != 0) return;
  uint32_t cum = 0, ab = 0; int bb = 0; bool found = false;
  for (int b = 255; b >= 0 && !found; --b) {
    uint32_t h = gh[c * 256 + b];
    if (cum + h >= (uint32_t)TOPK) { bb = b; ab = cum; found = true; }
    else cum += h;
  }
  if (!found) { bb = 0; ab = cum; }
  b1[c] = (uint32_t)bb; above1[c] = ab;
}

__global__ void k_scan2(const uint32_t* __restrict__ gh2, const uint32_t* __restrict__ b1,
                        const uint32_t* __restrict__ above1, uint32_t* __restrict__ thresh) {
  int c = blockIdx.x;
  if (threadIdx.x != 0) return;
  uint32_t cum = above1[c]; int b2 = 0; bool found = false;
  for (int b = 255; b >= 0 && !found; --b) {
    uint32_t h = gh2[c * 256 + b];
    if (cum + h >= (uint32_t)TOPK) { b2 = b; found = true; }
    else cum += h;
  }
  thresh[c] = ((b1[c] << 8) | (uint32_t)b2) << 16;
}

// ============= pass 3: compact candidates >= per-class threshold ======
__global__ void k_compact(const float* __restrict__ logits, const uint32_t* __restrict__ thresh,
                          uint32_t* __restrict__ cnt, unsigned long long* __restrict__ cand,
                          long long total) {
  uint32_t tk[N_CLASSES];
  for (int c = 0; c < N_CLASSES; ++c) tk[c] = thresh[c];
  long long stride = (long long)gridDim.x * blockDim.x;
  for (long long e = (long long)blockIdx.x * blockDim.x + threadIdx.x; e < total; e += stride) {
    __builtin_prefetch(logits + e + stride, 0, 1);
    uint32_t k = fkey(logits[e]);
    int c = (int)(e % N_CLASSES);
    if (k >= tk[c]) {
      uint32_t pos = atomicAdd(&cnt[c], 1u);
      if (pos < (uint32_t)CAND_CAP) {
        uint32_t aidx = (uint32_t)(e / N_CLASSES);
        // pack (key, ~idx): descending u64 sort == jax top_k order (ties -> lower idx)
        cand[(size_t)c * CAND_CAP + pos] =
            ((unsigned long long)k << 32) | (unsigned long long)(uint32_t)(~aidx);
      }
    }
  }
}

// ============= shared bitonic sort (descending) =======================
__device__ void bitonic_desc(unsigned long long* sh, int n) {
  for (int k = 2; k <= n; k <<= 1) {
    for (int j = k >> 1; j > 0; j >>= 1) {
      __syncthreads();
      for (int i = threadIdx.x; i < n; i += blockDim.x) {
        int p = i ^ j;
        if (p > i) {
          unsigned long long a = sh[i], b = sh[p];
          bool up = ((i & k) == 0);
          if (up ? (a < b) : (a > b)) { sh[i] = b; sh[p] = a; }
        }
      }
    }
  }
  __syncthreads();
}

// ============= pass 4: per-class exact ordered top-1000 ===============
__global__ void k_sortsel(const unsigned long long* __restrict__ cand,
                          const uint32_t* __restrict__ cnt,
                          uint32_t* __restrict__ selIdx, float* __restrict__ selSc) {
  __shared__ unsigned long long sh[CAND_CAP];   // 16 KB
  int c = blockIdx.x;
  uint32_t n = cnt[c]; if (n > (uint32_t)CAND_CAP) n = CAND_CAP;
  for (int i = threadIdx.x; i < CAND_CAP; i += blockDim.x)
    sh[i] = (i < (int)n) ? cand[(size_t)c * CAND_CAP + i] : 0ull;
  bitonic_desc(sh, CAND_CAP);
  for (int k = threadIdx.x; k < KPAD; k += blockDim.x) {
    if (k < TOPK && k < (int)n) {
      unsigned long long v = sh[k];
      uint32_t aidx = ~(uint32_t)v;
      float lg = unkey((uint32_t)(v >> 32));
      float sc = 1.0f / (1.0f + __expf(-lg));   // sigmoid only on survivors
      selIdx[c * KPAD + k] = aidx;
      selSc[c * KPAD + k]  = sc;
    } else {
      selIdx[c * KPAD + k] = 0u;
      selSc[c * KPAD + k]  = 0.0f;
    }
  }
}

// ============= pass 5: decode + clip + volume =========================
__global__ void k_decode(const float* __restrict__ anchors, const float* __restrict__ reg,
                         const uint32_t* __restrict__ selIdx, const float* __restrict__ selSc,
                         const int* __restrict__ imsz,
                         float* __restrict__ boxes, float* __restrict__ vol) {
  int tid = blockIdx.x * blockDim.x + threadIdx.x;
  if (tid >= N_CLASSES * KPAD) return;
  float hi = (float)(*imsz);
  uint32_t a = selIdx[tid];
  const float* A = anchors + (size_t)a * 6;
  const float* R = reg + (size_t)a * 6;
  float b[6]; float v = 1.0f;
  for (int d = 0; d < 3; ++d) {
    float w   = A[3 + d] - A[d];
    float ctr = A[d] + 0.5f * w;
    float pc  = R[d] * w + ctr;
    float ps  = __expf(fminf(R[3 + d], BBOX_CLIP)) * w;
    float lo  = fminf(fmaxf(pc - 0.5f * ps, 0.0f), hi);
    float hb  = fminf(fmaxf(pc + 0.5f * ps, 0.0f), hi);
    b[d] = lo; b[3 + d] = hb;
    v *= fmaxf(hb - lo, 0.0f);
  }
  bool live = selSc[tid] > 0.0f;
  for (int d = 0; d < 6; ++d) boxes[(size_t)tid * 6 + d] = live ? b[d] : 0.0f;
  vol[tid] = live ? v : 0.0f;
}

// ============= pass 6: suppression bitmask via WMMA volume-sum ========
// One wave32 per 16x16 tile. union(i,j) = (vol_i + vol_j) - inter;
// vol_i + vol_j computed as rank-2 outer product on the matrix core:
//   A(16x32 bf16) = [vol_i | 1 | 0 ...],  B(32x16 bf16) = [1 ; vol_j^T ; 0 ...]
__global__ void __launch_bounds__(32) k_supmat(const float* __restrict__ boxes,
                                               const float* __restrict__ vol,
                                               uint32_t* __restrict__ mask) {
  int tj = blockIdx.x, ti = blockIdx.y, c = blockIdx.z;
  int lane = threadIdx.x;
  int i0 = ti * 16, j0 = tj * 16;
  __shared__ float sbi[16][6], sbj[16][6], svi[16], svj[16];
  const float* bc = boxes + (size_t)c * KPAD * 6;
  for (int t = lane; t < 96; t += 32) {
    sbi[t / 6][t % 6] = bc[(i0 + t / 6) * 6 + t % 6];
    sbj[t / 6][t % 6] = bc[(j0 + t / 6) * 6 + t % 6];
  }
  if (lane < 16) svi[lane] = vol[c * KPAD + i0 + lane];
  else           svj[lane - 16] = vol[c * KPAD + j0 + (lane - 16)];
  __syncthreads();

  // Build A/B per the 16-bit A-matrix 16x32 VGPR layout:
  // lanes 0-15 hold K=0..7 in elements 0..7 (elem0=K0, elem1=K1); lanes 16-31 hold K=8.. -> zero.
  union { v16bf v; unsigned short s[16]; } ua, ub;
  for (int t = 0; t < 16; ++t) { ua.s[t] = 0; ub.s[t] = 0; }
  const unsigned short one_bf16 = (unsigned short)(__float_as_uint(1.0f) >> 16); // 0x3F80
  if (lane < 16) {
    ua.s[0] = (unsigned short)(__float_as_uint(svi[lane]) >> 16);  // A[m][0] = vol_i[m]
    ua.s[1] = one_bf16;                                            // A[m][1] = 1
    ub.s[0] = one_bf16;                                            // B[0][n] = 1
    ub.s[1] = (unsigned short)(__float_as_uint(svj[lane]) >> 16);  // B[1][n] = vol_j[n]
  }
  v8f acc = {};
  // D[m][n] = vol_i[m] + vol_j[n]   (emits v_wmma_f32_16x16x32_bf16)
  acc = __builtin_amdgcn_wmma_f32_16x16x32_bf16(false, ua.v, false, ub.v,
                                                (short)0, acc, false, false);

  int n    = lane & 15;        // column within tile (C/D layout: N = lane % 16)
  int half = lane >> 4;        // C/D layout: VGPR r -> M = r + 8*half
  uint32_t w     = (uint32_t)(j0 >> 5);
  uint32_t shift = (uint32_t)((tj & 1) * 16);
  for (int r = 0; r < 8; ++r) {
    int ri = r + 8 * half;
    float inter = 1.0f;
    for (int d = 0; d < 3; ++d) {
      float l  = fmaxf(sbi[ri][d],     sbj[n][d]);
      float rb = fminf(sbi[ri][3 + d], sbj[n][3 + d]);
      inter *= fmaxf(rb - l, 0.0f);
    }
    float un  = acc[r] - inter;                 // union from WMMA volume-sum
    float iou = inter / fmaxf(un, 1e-8f);
    unsigned long long bal = __ballot((int)(iou > NMS_TH));
    if (lane == 0) {
      uint32_t lo16 = (uint32_t)(bal & 0xffffull);          // row i0+r,   N=0..15
      uint32_t hi16 = (uint32_t)((bal >> 16) & 0xffffull);  // row i0+r+8, N=0..15
      if (lo16) atomicOr(&mask[((size_t)c * KPAD + (i0 + r))     * 32 + w], lo16 << shift);
      if (hi16) atomicOr(&mask[((size_t)c * KPAD + (i0 + r + 8)) * 32 + w], hi16 << shift);
    }
  }
}

// ============= pass 7: greedy NMS, one wave32 per class ===============
__global__ void __launch_bounds__(32) k_nms(const uint32_t* __restrict__ mask,
                                            const float* __restrict__ selSc,
                                            float* __restrict__ fsc) {
  int c = blockIdx.x, lane = threadIdx.x;
  const uint32_t* mrow = mask + (size_t)c * KPAD * 32;
  const float* sc = selSc + c * KPAD;
  uint32_t validW = 0;
  for (int b = 0; b < 32; ++b) {
    int i = 32 * lane + b;
    if (i < TOPK && sc[i] > SCORE_TH) validW |= (1u << b);
  }
  uint32_t removedW = 0, keepW = 0;
  uint32_t cur = mrow[lane];                          // row 0, pipelined
  for (int i = 0; i < TOPK; ++i) {
    uint32_t nxt = (i + 1 < TOPK) ? mrow[(size_t)(i + 1) * 32 + lane] : 0u;
    int wi = i >> 5, bi = i & 31;
    uint32_t rem = __shfl(removedW, wi, 32);
    uint32_t val = __shfl(validW,   wi, 32);
    bool keep = ((val >> bi) & 1u) && !((rem >> bi) & 1u);   // wave-uniform
    if (keep) {
      removedW |= cur;
      if (lane == wi) keepW |= (1u << bi);
    }
    cur = nxt;
  }
  for (int b = 0; b < 32; ++b) {
    int i = 32 * lane + b;
    fsc[c * KPAD + i] = ((keepW >> b) & 1u) ? sc[i] : 0.0f;
  }
}

// ============= pass 8: final top-300 over 8192 scores =================
__global__ void k_final(const float* __restrict__ fsc, const float* __restrict__ boxes,
                        float* __restrict__ out) {
  __shared__ unsigned long long sh[N_CLASSES * KPAD];   // 64 KB (<= 320 KB WGP LDS)
  for (int i = threadIdx.x; i < N_CLASSES * KPAD; i += blockDim.x) {
    float s = fsc[i];                                   // s >= 0 -> raw bits are ordered
    sh[i] = ((unsigned long long)__float_as_uint(s) << 32)
          | (unsigned long long)(uint32_t)(~(uint32_t)i);
  }
  bitonic_desc(sh, N_CLASSES * KPAD);
  for (int t = threadIdx.x; t < DET; t += blockDim.x) {
    unsigned long long v = sh[t];
    uint32_t fi = ~(uint32_t)v;
    float s = __uint_as_float((uint32_t)(v >> 32));
    const float* bp = boxes + (size_t)fi * 6;
    for (int d = 0; d < 6; ++d) out[t * 7 + d] = bp[d];
    out[t * 7 + 6] = s;
  }
}

// =====================================================================
extern "C" void kernel_launch(void* const* d_in, const int* in_sizes, int n_in,
                              void* d_out, int out_size, void* d_ws, size_t ws_size,
                              hipStream_t stream) {
  const float* anchors = (const float*)d_in[0];
  const float* boxreg  = (const float*)d_in[1];
  const float* logits  = (const float*)d_in[2];
  const int*   imsz    = (const int*)d_in[3];
  long long N = (long long)in_sizes[0] / 6;
  long long total = N * N_CLASSES;

  char* ws = (char*)d_ws;
  uint32_t* gh1    = (uint32_t*)(ws + WS_HIST1);
  uint32_t* gh2    = (uint32_t*)(ws + WS_HIST2);
  uint32_t* b1     = (uint32_t*)(ws + WS_B1);
  uint32_t* above1 = (uint32_t*)(ws + WS_ABOVE1);
  uint32_t* thresh = (uint32_t*)(ws + WS_THRESH);
  uint32_t* cnt    = (uint32_t*)(ws + WS_CNT);
  unsigned long long* cand = (unsigned long long*)(ws + WS_CAND);
  uint32_t* selIdx = (uint32_t*)(ws + WS_SELIDX);
  float*    selSc  = (float*)(ws + WS_SELSC);
  float*    boxes  = (float*)(ws + WS_BOXES);
  float*    vol    = (float*)(ws + WS_VOL);
  float*    fsc    = (float*)(ws + WS_FSC);
  uint32_t* mask   = (uint32_t*)(ws + WS_MASK);

  hipMemsetAsync(ws + WS_HIST1, 0, WS_ZERO1_BYTES, stream);   // hists + counters
  hipMemsetAsync(ws + WS_MASK,  0, WS_MASK_BYTES,  stream);   // suppression bits

  k_hist   <<<1024, 256, 0, stream>>>(logits, gh1, b1, 0, total);
  k_scan1  <<<N_CLASSES, 32, 0, stream>>>(gh1, b1, above1);
  k_hist   <<<1024, 256, 0, stream>>>(logits, gh2, b1, 1, total);
  k_scan2  <<<N_CLASSES, 32, 0, stream>>>(gh2, b1, above1, thresh);
  k_compact<<<1024, 256, 0, stream>>>(logits, thresh, cnt, cand, total);
  k_sortsel<<<N_CLASSES, 256, 0, stream>>>(cand, cnt, selIdx, selSc);
  k_decode <<<(N_CLASSES * KPAD + 255) / 256, 256, 0, stream>>>(anchors, boxreg, selIdx, selSc,
                                                                imsz, boxes, vol);
  dim3 gsup(KPAD / 16, KPAD / 16, N_CLASSES);   // 64 x 64 x 8 tiles, one wave each
  k_supmat <<<gsup, 32, 0, stream>>>(boxes, vol, mask);
  k_nms    <<<N_CLASSES, 32, 0, stream>>>(mask, selSc, fsc);
  k_final  <<<1, 1024, 0, stream>>>(fsc, boxes, (float*)d_out);
}
